// NodeClassifier_65506841199132
// MI455X (gfx1250) — matrix operand order repaired and checked by reading, not compiled, not run on hardware
//
#include <hip/hip_runtime.h>

typedef __attribute__((ext_vector_type(16))) _Float16 v16h;
typedef __attribute__((ext_vector_type(8)))  float    v8f;
typedef __attribute__((ext_vector_type(4)))  float    v4f;

#define F_IN 128
#define HPAD 64   // padded feature stride for all intermediate maps

__device__ __forceinline__ float relu_f(float x) { return x > 0.f ? x : 0.f; }

// ---------------------------------------------------------------------------
// Pre-pack a weight matrix W[fout_real, w_stride] (row-major f32) into WMMA
// A-fragment order, f16, zero-padded:  out[(ftile*nK + kc)*32 + lane][h],
// where half h of lane l holds A[m = ftile*16 + (l&15)]
//                              [k = kc*32 + ((l>>4)<<4) + h]
// (contiguous k order, matching the B-fragment fill).
// ---------------------------------------------------------------------------
__global__ void __launch_bounds__(32)
pack_weights(const float* __restrict__ W, _Float16* __restrict__ out,
             int w_stride, int k_real, int fout_real, int nK)
{
    const int lane  = threadIdx.x;
    const int kc    = blockIdx.x % nK;
    const int ftile = blockIdx.x / nK;
    const int lo = lane & 15, hi = lane >> 4;
    const int f  = ftile * 16 + lo;
    v16h a;
    #pragma unroll
    for (int h = 0; h < 16; ++h) {
        int k = kc * 32 + (hi << 4) + h;
        a[h] = (f < fout_real && k < k_real)
                 ? (_Float16)W[(size_t)f * w_stride + k] : (_Float16)0.f;
    }
    *(v16h*)(out + ((size_t)(ftile * nK + kc) * 32 + lane) * 16) = a;
}

__global__ void __launch_bounds__(64)
pack_bias(const float* __restrict__ b, float* __restrict__ out, int n_real)
{
    const int i = threadIdx.x;                 // 64 threads
    out[i] = (i < n_real) ? b[i] : 0.f;
}

// ---------------------------------------------------------------------------
// out[n, f] = act_out( sum_k act_in(in[n, k]) * W[f, k] + bias[f] )
// One wave = one 16x16 output tile; K fully unrolled (IN_STRIDE/32 WMMAs).
// Apk = pre-swizzled f16 A fragments, biasp = zero-padded bias.
// ---------------------------------------------------------------------------
template<int IN_STRIDE, int OUT_STRIDE, bool RELU_IN, bool RELU_OUT, bool FULL_STORE>
__global__ void __launch_bounds__(128)
gcn_wmma_gemm(const float* __restrict__ in, const _Float16* __restrict__ Apk,
              const float* __restrict__ biasp, float* __restrict__ out,
              int fout_guard)
{
    constexpr int NKC = IN_STRIDE / 32;
    const int lane  = threadIdx.x;              // 0..31
    const int lo    = lane & 15;
    const int hi    = lane >> 4;
    const int node  = blockIdx.x * 16 + lo;
    const int ftile = threadIdx.y;

    const float* bp = in + (size_t)node * IN_STRIDE + hi * 16;
    const v16h*  ap = (const v16h*)Apk + (size_t)ftile * NKC * 32 + lane;

    v8f acc = {};
    #pragma unroll
    for (int kc = 0; kc < NKC; ++kc) {
        // B tile (activations) 32x16: lane holds node column, 16 contiguous k
        v16h b;
        #pragma unroll
        for (int q = 0; q < 4; ++q) {
            v4f t = *(const v4f*)(bp + kc * 32 + q * 4);
            if (RELU_IN) {
                t.x = relu_f(t.x); t.y = relu_f(t.y);
                t.z = relu_f(t.z); t.w = relu_f(t.w);
            }
            b[q * 4 + 0] = (_Float16)t.x;
            b[q * 4 + 1] = (_Float16)t.y;
            b[q * 4 + 2] = (_Float16)t.z;
            b[q * 4 + 3] = (_Float16)t.w;
        }
        // A tile (pre-swizzled weights): one 32B load
        v16h a = ap[kc * 32];
        acc = __builtin_amdgcn_wmma_f32_16x16x32_f16(
                  false, a, false, b, (short)0, acc, false, false);
    }

    // Epilogue: D VGPR r -> (f = ftile*16 + hi*8 + r, n = node)
    const int fbase = ftile * 16 + hi * 8;
    float* op = out + (size_t)node * OUT_STRIDE + fbase;
    v4f b0 = *(const v4f*)(biasp + fbase);
    v4f b1 = *(const v4f*)(biasp + fbase + 4);

    float vv[8];
    #pragma unroll
    for (int r = 0; r < 8; ++r) {
        float bia = (r < 4) ? ((r==0)?b0.x:(r==1)?b0.y:(r==2)?b0.z:b0.w)
                            : ((r==4)?b1.x:(r==5)?b1.y:(r==6)?b1.z:b1.w);
        float v = acc[r] + bia;
        if (RELU_OUT) v = relu_f(v);
        vv[r] = v;
    }

    if constexpr (FULL_STORE) {
        v4f s0 = {vv[0], vv[1], vv[2], vv[3]};
        v4f s1 = {vv[4], vv[5], vv[6], vv[7]};
        *(v4f*)op       = s0;
        *(v4f*)(op + 4) = s1;
    } else {
        #pragma unroll
        for (int r = 0; r < 8; ++r)
            if (fbase + r < fout_guard) op[r] = vv[r];
    }
}

// ---------------------------------------------------------------------------
// agg[dst] += h[src]; one wave per edge, 128B coalesced reads and contiguous
// global f32 atomic adds (L2-resident: feature maps are 25.6 MB << 192 MB L2).
// ---------------------------------------------------------------------------
__global__ void __launch_bounds__(256)
gcn_scatter_add(const float* __restrict__ h, float* __restrict__ agg,
                const int* __restrict__ src, const int* __restrict__ dst,
                int num_edges, int stride, int nfeat)
{
    const int lane = threadIdx.x & 31;
    const int e = (int)((blockIdx.x * blockDim.x + threadIdx.x) >> 5);
    if (e >= num_edges) return;
    const int s = src[e];
    const int d = dst[e];
    const float* hp = h + (size_t)s * stride;
    float*       ap = agg + (size_t)d * stride;
    for (int f = lane; f < nfeat; f += 32)
        unsafeAtomicAdd(ap + f, hp[f]);
}

extern "C" void kernel_launch(void* const* d_in, const int* in_sizes, int n_in,
                              void* d_out, int out_size, void* d_ws, size_t ws_size,
                              hipStream_t stream) {
    const float* x      = (const float*)d_in[0];
    const int*   ei     = (const int*)  d_in[1];
    const float* w1_lin = (const float*)d_in[2];
    const float* b1_lin = (const float*)d_in[3];
    const float* w1_o1  = (const float*)d_in[4];
    const float* b1_o1  = (const float*)d_in[5];
    const float* w1_o2  = (const float*)d_in[6];
    const float* b1_o2  = (const float*)d_in[7];
    const float* w2_lin = (const float*)d_in[8];
    const float* b2_lin = (const float*)d_in[9];
    const float* w2_o1  = (const float*)d_in[10];
    const float* b2_o1  = (const float*)d_in[11];
    const float* w2_o2  = (const float*)d_in[12];
    const float* b2_o2  = (const float*)d_in[13];

    const int Nn = in_sizes[0] / F_IN;   // 100000 (divisible by 16)
    const int E  = in_sizes[1] / 2;      // 1000000
    const int* src = ei;
    const int* dst = ei + E;

    // ---- workspace layout ----
    const size_t per = (size_t)Nn * HPAD;            // floats per feature map
    float* ws0 = (float*)d_ws;
    float* ws1 = ws0 + per;
    float* ws2 = ws1 + per;
    float* wsb = ws2 + per;                           // 6 padded biases x 64
    float* bb1_lin = wsb + 0 * 64;
    float* bb1_o1  = wsb + 1 * 64;
    float* bb1_o2  = wsb + 2 * 64;
    float* bb2_lin = wsb + 3 * 64;
    float* bb2_o1  = wsb + 4 * 64;
    float* bb2_o2  = wsb + 5 * 64;
    _Float16* wsh = (_Float16*)(wsb + 6 * 64);        // packed weight fragments
    _Float16* pw1_lin = wsh;                          // 4 tiles x 4 kc
    _Float16* pw1_o1  = pw1_lin + 16 * 32 * 16;       // 4 x 2
    _Float16* pw1_o2  = pw1_o1  +  8 * 32 * 16;
    _Float16* pw2_lin = pw1_o2  +  8 * 32 * 16;
    _Float16* pw2_o1  = pw2_lin +  8 * 32 * 16;
    _Float16* pw2_o2  = pw2_o1  +  8 * 32 * 16;       // 3 x 2

    // ---- pack weights (once per launch; trivially cheap) ----
    pack_weights<<<16, 32, 0, stream>>>(w1_lin, pw1_lin, 128, 128, 64, 4);
    pack_weights<<< 8, 32, 0, stream>>>(w1_o1,  pw1_o1,   64,  64, 64, 2);
    pack_weights<<< 8, 32, 0, stream>>>(w1_o2,  pw1_o2,   64,  64, 64, 2);
    pack_weights<<< 8, 32, 0, stream>>>(w2_lin, pw2_lin,  64,  64, 40, 2);
    pack_weights<<< 8, 32, 0, stream>>>(w2_o1,  pw2_o1,   40,  40, 40, 2);
    pack_weights<<< 6, 32, 0, stream>>>(w2_o2,  pw2_o2,   40,  40, 40, 2);
    pack_bias<<<1, 64, 0, stream>>>(b1_lin, bb1_lin, 64);
    pack_bias<<<1, 64, 0, stream>>>(b1_o1,  bb1_o1,  64);
    pack_bias<<<1, 64, 0, stream>>>(b1_o2,  bb1_o2,  64);
    pack_bias<<<1, 64, 0, stream>>>(b2_lin, bb2_lin, 40);
    pack_bias<<<1, 64, 0, stream>>>(b2_o1,  bb2_o1,  40);
    pack_bias<<<1, 64, 0, stream>>>(b2_o2,  bb2_o2,  40);

    dim3 gN(Nn / 16);
    dim3 blk4(32, 4);                     // 64-wide padded outputs
    dim3 blk3(32, 3);                     // 40-wide packed final output
    const int edgeBlocks = (E + 7) / 8;   // 8 edges per 256-thread block

    // ---- conv1 ----
    gcn_wmma_gemm<128, 64, false, false, true>
        <<<gN, blk4, 0, stream>>>(x, pw1_lin, bb1_lin, ws0, 64);
    hipMemcpyAsync(ws1, ws0, per * sizeof(float), hipMemcpyDeviceToDevice, stream);
    gcn_scatter_add<<<edgeBlocks, 256, 0, stream>>>(ws0, ws1, src, dst, E, 64, 64);
    gcn_wmma_gemm<64, 64, true, true, true>
        <<<gN, blk4, 0, stream>>>(ws1, pw1_o1, bb1_o1, ws2, 64);
    gcn_wmma_gemm<64, 64, false, true, true>
        <<<gN, blk4, 0, stream>>>(ws2, pw1_o2, bb1_o2, ws0, 64);

    // ---- conv2 ----
    gcn_wmma_gemm<64, 64, false, false, true>
        <<<gN, blk4, 0, stream>>>(ws0, pw2_lin, bb2_lin, ws1, 64);
    hipMemcpyAsync(ws2, ws1, per * sizeof(float), hipMemcpyDeviceToDevice, stream);
    gcn_scatter_add<<<edgeBlocks, 256, 0, stream>>>(ws1, ws2, src, dst, E, 64, 40);
    gcn_wmma_gemm<64, 64, true, true, true>
        <<<gN, blk4, 0, stream>>>(ws2, pw2_o1, bb2_o1, ws0, 64);
    gcn_wmma_gemm<64, 40, false, false, false>
        <<<gN, blk3, 0, stream>>>(ws0, pw2_o2, bb2_o2, (float*)d_out, 40);
}
